// PairInitProjection_68384469286922
// MI455X (gfx1250) — compile-verified, matrix operand order checked by reading, not run on hardware
//
#include <hip/hip_runtime.h>
#include <hip/hip_bf16.h>
#include <stdint.h>

typedef __attribute__((ext_vector_type(16))) _Float16 v16h;
typedef __attribute__((ext_vector_type(8)))  float    v8f;

#define NA    1024
#define NB    1024
#define DSG   512
#define RANK  32
#define DP    128
#define NRBF  64
#define LWP   40   // padded LDS pitch (halves) for lw rows  -> conflict-free b128
#define WDP   72   // padded LDS pitch (halves) for W_dist rows

union V16 { v16h v; _Float16 h[16]; uint4 q[2]; };
union V8  { v8f  v; float f[8]; };

// ---------------- prep: left (f32) and right (f16) projections ----------------
__global__ void k_proj(const float* __restrict__ sa, const float* __restrict__ sb,
                       const float* __restrict__ Wl, const float* __restrict__ bl,
                       const float* __restrict__ Wr, const float* __restrict__ br,
                       float* __restrict__ left_f, _Float16* __restrict__ right_h) {
  int idx = blockIdx.x * blockDim.x + threadIdx.x;           // 65536 total
  bool is_left = idx < NA * RANK;
  int li = is_left ? idx : idx - NA * RANK;
  int i = li >> 5, r = li & 31;
  const float4* arow = (const float4*)((is_left ? sa : sb) + (size_t)i * DSG);
  const float4* wrow = (const float4*)((is_left ? Wl : Wr) + (size_t)r * DSG);
  float acc = 0.f;
  #pragma unroll 4
  for (int k = 0; k < DSG / 4; ++k) {
    float4 a = arow[k], w = wrow[k];
    acc += a.x * w.x + a.y * w.y + a.z * w.z + a.w * w.w;
  }
  acc += is_left ? bl[r] : br[r];
  if (is_left) left_f[li] = acc; else right_h[li] = (_Float16)acc;
}

// ---------------- prep: lw[i,p,s] = sum_r left[i,r] * W_outer[p, r*32+s] (f16) -
__global__ void k_lw(const float* __restrict__ left_f, const float* __restrict__ Wo,
                     _Float16* __restrict__ lw_h) {
  int idx = blockIdx.x * blockDim.x + threadIdx.x;           // NA*DP
  int i = idx >> 7, p = idx & 127;
  float acc[RANK];
  #pragma unroll
  for (int s = 0; s < RANK; ++s) acc[s] = 0.f;
  const float* wrow = Wo + (size_t)p * RANK * RANK;
  const float* lrow = left_f + (size_t)i * RANK;
  for (int r = 0; r < RANK; ++r) {
    float lv = lrow[r];
    const float4* w4 = (const float4*)(wrow + r * RANK);
    #pragma unroll
    for (int s4 = 0; s4 < RANK / 4; ++s4) {
      float4 w = w4[s4];
      acc[s4*4+0] += lv * w.x; acc[s4*4+1] += lv * w.y;
      acc[s4*4+2] += lv * w.z; acc[s4*4+3] += lv * w.w;
    }
  }
  _Float16* dst = lw_h + (size_t)idx * RANK;
  #pragma unroll
  for (int s = 0; s < RANK; ++s) dst[s] = (_Float16)acc[s];
}

// ---------------- prep: W_dist f32 -> f16 ----------------
__global__ void k_cvt(const float* __restrict__ src, _Float16* __restrict__ dst) {
  int idx = blockIdx.x * blockDim.x + threadIdx.x;           // 8192
  dst[idx] = (_Float16)src[idx];
}

// ---------------- fused pair kernel --------------------------------------------
// Block: 128 threads (4 waves). Block -> one i. Wave -> j-tiles {w, w+4, ...}.
// Per (i, j16) tile: 8 p-tiles; each = 1 WMMA (lw·right, K=32) + 2 WMMA (Wd·rbf, K=64).
__global__ __launch_bounds__(128)
void k_pair(const _Float16* __restrict__ right_h, const _Float16* __restrict__ lw_h,
            const _Float16* __restrict__ wd_h,
            const float* __restrict__ x_a, const float* __restrict__ x_b,
            const float* __restrict__ b_outer, const float* __restrict__ b_dist,
            const float* __restrict__ gamma, const float* __restrict__ beta,
            const float* __restrict__ centers, float* __restrict__ out) {
  __shared__ __align__(16) _Float16 lw_s[DP * LWP];   // 10240 B, padded rows
  __shared__ __align__(16) _Float16 wd_s[DP * WDP];   // 18432 B, padded rows
  __shared__ float cent_s[NRBF];
  __shared__ float bsum_s[DP], g_s[DP], be_s[DP];

  const int tid = threadIdx.x;
  const int i   = blockIdx.x;

  // ---- stage LDS: one row per thread (128 rows each) ----
  {
    const uint32_t* s = (const uint32_t*)(lw_h + (size_t)i * DP * RANK) + tid * 16;
    uint32_t* d = (uint32_t*)lw_s + tid * (LWP / 2);
    #pragma unroll
    for (int t = 0; t < 16; ++t) d[t] = s[t];
    const uint32_t* s2 = (const uint32_t*)wd_h + tid * 32;
    uint32_t* d2 = (uint32_t*)wd_s + tid * (WDP / 2);
    #pragma unroll
    for (int t = 0; t < 32; ++t) d2[t] = s2[t];
    if (tid < NRBF) cent_s[tid] = centers[tid];
    if (tid < DP) { bsum_s[tid] = b_outer[tid] + b_dist[tid]; g_s[tid] = gamma[tid]; be_s[tid] = beta[tid]; }
  }
  __syncthreads();

  const int lane = tid & 31;
  const int wave = tid >> 5;
  const int col  = lane & 15;                 // p column within tile / j row index
  const int s0   = (lane < 16) ? 0 : 8;       // per-lane K-chunk base (16b A/B layout)
  const float xa0 = x_a[i*3+0], xa1 = x_a[i*3+1], xa2 = x_a[i*3+2];
  const float inv_sigma = 10.24f;             // 1 / (20/64)^2

  #pragma unroll 1
  for (int jt = wave; jt < NB / 16; jt += 4) {
    const int j0 = jt * 16;
    const int j  = j0 + col;

    // A1: right[j, 0..31] in 16-bit A layout (two 16B chunks at s0, s0+16)
    V16 aR;
    const _Float16* rrow = right_h + (size_t)j * RANK;
    aR.q[0] = *(const uint4*)(rrow + s0);
    aR.q[1] = *(const uint4*)(rrow + s0 + 16);
    if (jt + 4 < NB / 16)
      __builtin_prefetch(right_h + (size_t)(j + 64) * RANK, 0, 3);

    // distance + RBF A operands (16x64 split into two K=32 chunks)
    float dx = xa0 - x_b[j*3+0], dy = xa1 - x_b[j*3+1], dz = xa2 - x_b[j*3+2];
    float d = __builtin_amdgcn_sqrtf(dx*dx + dy*dy + dz*dz);   // raw v_sqrt_f32 (TRANS)
    V16 aE[2];
    #pragma unroll
    for (int kc = 0; kc < 2; ++kc) {
      #pragma unroll
      for (int u = 0; u < 16; ++u) {
        int k = kc*32 + s0 + ((u < 8) ? u : (8 + u));   // halves: {s0..s0+7, s0+16..s0+23}
        float t = d - cent_s[k];
        aE[kc].h[u] = (_Float16)__expf(-t * t * inv_sigma);
      }
    }

    V8 acc[8];
    #pragma unroll
    for (int t8 = 0; t8 < 8; ++t8) {
      #pragma unroll
      for (int e = 0; e < 8; ++e) acc[t8].f[e] = 0.f;
    }

    #pragma unroll
    for (int t8 = 0; t8 < 8; ++t8) {
      const int p = t8 * 16 + col;
      V16 b1, b2a, b2b;
      b1.q[0]  = *(const uint4*)&lw_s[p * LWP + s0];
      b1.q[1]  = *(const uint4*)&lw_s[p * LWP + s0 + 16];
      b2a.q[0] = *(const uint4*)&wd_s[p * WDP + s0];
      b2a.q[1] = *(const uint4*)&wd_s[p * WDP + s0 + 16];
      b2b.q[0] = *(const uint4*)&wd_s[p * WDP + 32 + s0];
      b2b.q[1] = *(const uint4*)&wd_s[p * WDP + 32 + s0 + 16];
      acc[t8].v = __builtin_amdgcn_wmma_f32_16x16x32_f16(false, aR.v,    false, b1.v,  (short)0, acc[t8].v, false, false);
      acc[t8].v = __builtin_amdgcn_wmma_f32_16x16x32_f16(false, aE[0].v, false, b2a.v, (short)0, acc[t8].v, false, false);
      acc[t8].v = __builtin_amdgcn_wmma_f32_16x16x32_f16(false, aE[1].v, false, b2b.v, (short)0, acc[t8].v, false, false);
    }

    // bias (b_outer + b_dist), per p column
    #pragma unroll
    for (int t8 = 0; t8 < 8; ++t8) {
      float bs = bsum_s[t8 * 16 + col];
      #pragma unroll
      for (int e = 0; e < 8; ++e) acc[t8].f[e] += bs;
    }

    // LayerNorm over p (lane dim within 16-lane half) + affine + store
    const int jbase = (lane >= 16) ? 8 : 0;
    #pragma unroll
    for (int rr = 0; rr < 8; ++rr) {
      float s1 = 0.f, s2 = 0.f;
      #pragma unroll
      for (int t8 = 0; t8 < 8; ++t8) { float x = acc[t8].f[rr]; s1 += x; s2 += x * x; }
      #pragma unroll
      for (int m = 1; m < 16; m <<= 1) {
        s1 += __shfl_xor(s1, m, 32);
        s2 += __shfl_xor(s2, m, 32);
      }
      float mean = s1 * (1.f / 128.f);
      float var  = s2 * (1.f / 128.f) - mean * mean;
      float inv  = __builtin_amdgcn_rsqf(var + 1e-5f);   // raw v_rsq_f32 (TRANS)
      float* orow = out + ((size_t)i * NB + (size_t)(j0 + rr + jbase)) * DP;
      #pragma unroll
      for (int t8 = 0; t8 < 8; ++t8) {
        int p = t8 * 16 + col;
        orow[p] = (acc[t8].f[rr] - mean) * inv * g_s[p] + be_s[p];
      }
    }
  }
}

extern "C" void kernel_launch(void* const* d_in, const int* in_sizes, int n_in,
                              void* d_out, int out_size, void* d_ws, size_t ws_size,
                              hipStream_t stream) {
  (void)in_sizes; (void)n_in; (void)out_size; (void)ws_size;
  const float* sa = (const float*)d_in[0];
  const float* sb = (const float*)d_in[1];
  const float* xa = (const float*)d_in[2];
  const float* xb = (const float*)d_in[3];
  const float* Wl = (const float*)d_in[4];
  const float* bl = (const float*)d_in[5];
  const float* Wr = (const float*)d_in[6];
  const float* br = (const float*)d_in[7];
  const float* Wo = (const float*)d_in[8];
  const float* bo = (const float*)d_in[9];
  const float* Wd = (const float*)d_in[10];
  const float* bd = (const float*)d_in[11];
  const float* gm = (const float*)d_in[12];
  const float* bt = (const float*)d_in[13];
  const float* ct = (const float*)d_in[14];
  float* out = (float*)d_out;

  char* ws = (char*)d_ws;
  float*    left_f  = (float*)ws;                                  // 128 KB
  _Float16* right_h = (_Float16*)(ws + 131072);                    //  64 KB
  _Float16* wd_h    = (_Float16*)(ws + 131072 + 65536);            //  16 KB
  _Float16* lw_h    = (_Float16*)(ws + 131072 + 65536 + 16384);    //   8 MB

  k_proj<<<256, 256, 0, stream>>>(sa, sb, Wl, bl, Wr, br, left_f, right_h);
  k_cvt <<<32,  256, 0, stream>>>(Wd, wd_h);
  k_lw  <<<1024, 128, 0, stream>>>(left_f, Wo, lw_h);
  k_pair<<<NA, 128, 0, stream>>>(right_h, lw_h, wd_h, xa, xb, bo, bd, gm, bt, ct, out);
}
